// PyGraphBipartite_56143812493353
// MI455X (gfx1250) — matrix operand order repaired and checked by reading, not compiled, not run on hardware
//
#include <hip/hip_runtime.h>
#include <hip/hip_bf16.h>

// ---------------------------------------------------------------------------
// Problem constants (B=8, C=64, H=W=32 -> N=8192, OUT=64, K=9 neighbors)
// ---------------------------------------------------------------------------
#define NNODES   8192
#define NCH      64
#define NBATCH   8            // batches 0..7 contiguous; node 8191 is singleton batch 8
#define BATCHCAP 1024         // batches 0..6 have 1024 nodes, batch 7 has 1023
#define KNN      9

typedef __attribute__((ext_vector_type(16))) _Float16 v16h;
typedef __attribute__((ext_vector_type(8)))  _Float16 v8h;
typedef __attribute__((ext_vector_type(8)))  float    v8f;

static __device__ __forceinline__ int imin(int a, int b) { return a < b ? a : b; }

// ---------------------------------------------------------------------------
// Kernel 1: BCHW -> [N, C] transpose through padded LDS tile (conflict-free)
// grid 128, block 256. Each block: one batch b, 64 consecutive hw, all 64 ch.
// ---------------------------------------------------------------------------
__global__ void transpose_kernel(const float* __restrict__ x, float* __restrict__ xf) {
    __shared__ float tile[64][65];
    int blk   = blockIdx.x;
    int b     = blk >> 4;          // 16 blocks per image
    int hw0   = (blk & 15) * 64;
    int t     = threadIdx.x;
    for (int i = t; i < 4096; i += 256) {
        int c = i >> 6, w = i & 63;
        tile[c][w] = x[(b * NCH + c) * 1024 + hw0 + w];       // coalesced over hw
    }
    __syncthreads();
    int n0 = b * 1024 + hw0;
    for (int i = t; i < 4096; i += 256) {
        int nl = i >> 6, c = i & 63;
        xf[(n0 + nl) * NCH + c] = tile[c][nl];                // coalesced over c
    }
}

// ---------------------------------------------------------------------------
// Kernel 2: fp16 copy + squared norms. One wave32 per node; lane handles
// channels (lane, lane+32); shuffle tree reduction (wave32 semantics).
// grid 1024, block 256 (8 waves).
// ---------------------------------------------------------------------------
__global__ void norm_kernel(const float* __restrict__ xf,
                            _Float16* __restrict__ xh,
                            float* __restrict__ sqn) {
    int gtid = blockIdx.x * blockDim.x + threadIdx.x;
    int node = gtid >> 5;
    int lane = threadIdx.x & 31;
    float v0 = xf[node * NCH + lane];
    float v1 = xf[node * NCH + lane + 32];
    xh[node * NCH + lane]      = (_Float16)v0;
    xh[node * NCH + lane + 32] = (_Float16)v1;
    float s = v0 * v0 + v1 * v1;
    #pragma unroll
    for (int off = 16; off > 0; off >>= 1) s += __shfl_xor(s, off, 32);
    if (lane == 0) sqn[node] = s;
}

// ---------------------------------------------------------------------------
// WMMA fragment loaders for V_WMMA_F32_16X16X32_F16 (wave32 layouts, ISA 7.12.2)
// A (16x32, MxK): lane m(0-15): K{0..7,16..23}; lane m+16: K{8..15,24..31}
// B (32x16, KxN): lane n(0-15): K{0..7,16..23}; lane n+16: K{8..15,24..31}
// ---------------------------------------------------------------------------
static __device__ __forceinline__ v16h load_a_frag(const _Float16* __restrict__ xh,
                                                   int row, int koff, int lane) {
    int kbase = koff + ((lane & 16) ? 8 : 0);
    const _Float16* p = xh + row * NCH + kbase;
    v8h lo = *(const v8h*)(p);            // K .. K+7
    v8h hi = *(const v8h*)(p + 16);       // K+16 .. K+23
    return __builtin_shufflevector(lo, hi, 0,1,2,3,4,5,6,7,8,9,10,11,12,13,14,15);
}

static __device__ __forceinline__ v16h load_b_frag(const _Float16* __restrict__ xh,
                                                   int col_node, int koff, int lane) {
    int kbase = koff + ((lane & 16) ? 8 : 0);
    const _Float16* p = xh + col_node * NCH + kbase;
    v8h lo = *(const v8h*)(p);
    v8h hi = *(const v8h*)(p + 16);
    return __builtin_shufflevector(lo, hi, 0,1,2,3,4,5,6,7,8,9,10,11,12,13,14,15);
}

// ---------------------------------------------------------------------------
// Kernel 3: per-batch kNN + neighbor mean.
// grid 512 (8 batches x 64 i-tiles of 16 rows), block 256 (8 waves),
// dynamic LDS = 16*1024 floats (64 KB distance stripe; needs CDNA5 320KB WGP LDS).
// Phase 1: Gram via v_wmma_f32_16x16x32_f16 (A reused over 8 j-tiles/wave).
// Phase 2: 9x argmin sweep per row (shuffle reduce, knock out winner with +inf).
// Phase 3: gather the 9 neighbor rows from fp32 x_f, mean, store.
// ---------------------------------------------------------------------------
__global__ void knn_mean_kernel(const float* __restrict__ xf,
                                const _Float16* __restrict__ xh,
                                const float* __restrict__ sqn,
                                float* __restrict__ mean) {
    extern __shared__ float dist_s[];          // [16][1024]
    __shared__ float s_sqni[16];
    __shared__ int   s_idx[16][KNN];

    const float INF = __builtin_huge_valf();
    int tid  = threadIdx.x;
    int lane = tid & 31;
    int wave = tid >> 5;

    int blk    = blockIdx.x;
    int b      = blk >> 6;                     // batch 0..7
    int tile   = blk & 63;                     // i-tile within batch
    int bstart = b * BATCHCAP;
    int bsize  = (b == 7) ? (BATCHCAP - 1) : BATCHCAP;
    int i0     = bstart + tile * 16;
    int rows   = imin(16, bstart + bsize - i0);    // 15 on the last tile of batch 7
    int jlast  = bstart + bsize - 1;

    if (tid < 16) s_sqni[tid] = sqn[imin(i0 + tid, jlast)];
    __syncthreads();

    // ----- Phase 1: distance stripe via WMMA -----
    int mrow = lane & 15;
    int ia   = i0 + imin(mrow, rows - 1);
    v16h a0 = load_a_frag(xh, ia, 0,  lane);   // K = 0..31
    v16h a1 = load_a_frag(xh, ia, 32, lane);   // K = 32..63

    int   mbase = (lane & 16) ? 8 : 0;
    float sq_i[8];
    #pragma unroll
    for (int r = 0; r < 8; ++r) sq_i[r] = s_sqni[mbase + r];

    for (int jt = wave; jt < 64; jt += 8) {
        int j0 = bstart + jt * 16;
        int jn = j0 + (lane & 15);
        int jc = imin(jn, jlast);
        __builtin_prefetch(xh + imin(jc + 128, jlast) * NCH, 0, 1);  // global_prefetch_b8

        v16h b0 = load_b_frag(xh, jc, 0,  lane);
        v16h b1 = load_b_frag(xh, jc, 32, lane);

        v8f acc = {};
        acc = __builtin_amdgcn_wmma_f32_16x16x32_f16(false, a0, false, b0,
                                                     (short)0, acc, false, false);
        acc = __builtin_amdgcn_wmma_f32_16x16x32_f16(false, a1, false, b1,
                                                     (short)0, acc, false, false);

        float sj  = sqn[jc];
        int   col = jt * 16 + (lane & 15);
        bool  ok  = col < bsize;
        #pragma unroll
        for (int r = 0; r < 8; ++r) {
            float d = ok ? (sq_i[r] + sj - 2.0f * acc[r]) : INF;
            dist_s[(mbase + r) * BATCHCAP + col] = d;
        }
    }
    __syncthreads();

    // ----- Phase 2: 9 nearest per row -----
    for (int m = wave; m < 16; m += 8) {
        if (m < rows) {
            for (int q = 0; q < KNN; ++q) {
                float best = INF; int bidx = 0;
                for (int col = lane; col < BATCHCAP; col += 32) {
                    float v = dist_s[m * BATCHCAP + col];
                    if (v < best) { best = v; bidx = col; }
                }
                #pragma unroll
                for (int off = 16; off > 0; off >>= 1) {
                    float ov = __shfl_xor(best, off, 32);
                    int   oi = __shfl_xor(bidx, off, 32);
                    if (ov < best) { best = ov; bidx = oi; }
                }
                bidx = __shfl(bidx, 0, 32);
                if (lane == 0) {
                    s_idx[m][q] = bstart + bidx;
                    dist_s[m * BATCHCAP + bidx] = INF;   // knock out for next sweep
                }
                __builtin_amdgcn_wave_barrier();
            }
        }
    }
    __syncthreads();

    // ----- Phase 3: mean of 9 neighbor feature rows (fp32 exact) -----
    for (int m = wave; m < 16; m += 8) {
        if (m < rows) {
            #pragma unroll
            for (int cc = 0; cc < 2; ++cc) {
                int c = lane + cc * 32;
                float s = 0.0f;
                #pragma unroll
                for (int q = 0; q < KNN; ++q) s += xf[s_idx[m][q] * NCH + c];
                mean[(i0 + m) * NCH + c] = s * (1.0f / 9.0f);
            }
        }
    }

    // singleton node 8191 (batch 8): only neighbor is itself
    if (blk == 511 && wave == 0) {
        #pragma unroll
        for (int cc = 0; cc < 2; ++cc) {
            int c = lane + cc * 32;
            mean[(NNODES - 1) * NCH + c] = xf[(NNODES - 1) * NCH + c];
        }
    }
}

// ---------------------------------------------------------------------------
// Kernel 4: out = mean @ W_l^T + b_l + x_f @ W_r^T   (fp32 exact)
// grid 512, block 256; each block does 16 nodes, weights cached in LDS with
// pitch 65 (conflict-free: bank = (o+c) % 64 distinct across a wave).
// ---------------------------------------------------------------------------
__global__ void sage_out_kernel(const float* __restrict__ xf,
                                const float* __restrict__ mean,
                                const float* __restrict__ Wl,
                                const float* __restrict__ bl,
                                const float* __restrict__ Wr,
                                float* __restrict__ out) {
    __shared__ float wl[64 * 65];
    __shared__ float wr[64 * 65];
    __shared__ float sx[4][64];
    __shared__ float sm[4][64];
    int t = threadIdx.x;
    for (int i = t; i < 4096; i += 256) {
        int o = i >> 6, c = i & 63;
        wl[o * 65 + c] = Wl[i];
        wr[o * 65 + c] = Wr[i];
    }
    __syncthreads();

    int n0  = blockIdx.x * 16;
    int sub = t >> 6;        // 0..3 concurrent nodes
    int o   = t & 63;        // output channel
    float bias = bl[o];

    for (int chunk = 0; chunk < 4; ++chunk) {
        int node = n0 + chunk * 4 + sub;
        sx[sub][o] = xf[node * NCH + o];
        sm[sub][o] = mean[node * NCH + o];
        __syncthreads();
        float acc = bias;
        #pragma unroll 8
        for (int c = 0; c < 64; ++c)
            acc += sm[sub][c] * wl[o * 65 + c] + sx[sub][c] * wr[o * 65 + c];
        out[node * NCH + o] = acc;
        __syncthreads();
    }
}

// ---------------------------------------------------------------------------
// Launch. Inputs: x[8,64,32,32], W_l[64,64], b_l[64], W_r[64,64]; out f32 [8192,64]
// Workspace: xf 2MB | xh 1MB | sqn 32KB | mean 2MB  (~5.2 MB total)
// ---------------------------------------------------------------------------
extern "C" void kernel_launch(void* const* d_in, const int* in_sizes, int n_in,
                              void* d_out, int out_size, void* d_ws, size_t ws_size,
                              hipStream_t stream) {
    const float* x  = (const float*)d_in[0];
    const float* Wl = (const float*)d_in[1];
    const float* bl = (const float*)d_in[2];
    const float* Wr = (const float*)d_in[3];
    float* out = (float*)d_out;

    char* ws = (char*)d_ws;
    float*    xf   = (float*)(ws);
    _Float16* xh   = (_Float16*)(ws + (size_t)2 * 1024 * 1024);
    float*    sqn  = (float*)(ws + (size_t)3 * 1024 * 1024);
    float*    mean = (float*)(ws + (size_t)3 * 1024 * 1024 + 64 * 1024);

    transpose_kernel<<<128, 256, 0, stream>>>(x, xf);
    norm_kernel<<<1024, 256, 0, stream>>>(xf, xh, sqn);
    knn_mean_kernel<<<512, 256, 16 * BATCHCAP * sizeof(float), stream>>>(xf, xh, sqn, mean);
    sage_out_kernel<<<512, 256, 0, stream>>>(xf, mean, Wl, bl, Wr, out);
}
